// MultiQueryAttention_1769526526610
// MI455X (gfx1250) — matrix operand order repaired and checked by reading, not compiled, not run on hardware
//
#include <hip/hip_runtime.h>
#include <hip/hip_bf16.h>

// MQA forward for MI455X (gfx1250, wave32, WMMA).
// Pipeline: convert/transpose -> WMMA GEMMs (Q,K,V) -> flash-attention (WMMA)
// -> WMMA GEMM with W_o. All matmuls use v_wmma_f32_16x16x32_f16 (f32 accum).
// GEMM wave tile is 32x64 (8 WMMAs per 12 b128 loads per k-step of 32).

#define DMODEL 2048
#define NHEADS 16
#define DK     128
#define BATCHN 2
#define SEQN   2048

typedef __attribute__((ext_vector_type(16))) _Float16 v16h;
typedef __attribute__((ext_vector_type(8)))  _Float16 v8h;
typedef __attribute__((ext_vector_type(8)))  float    v8f;

__device__ __forceinline__ v16h mkv16(v8h lo, v8h hi) {
  v16h r;
#pragma unroll
  for (int i = 0; i < 8; ++i) { r[i] = lo[i]; r[i + 8] = hi[i]; }
  return r;
}

__device__ __forceinline__ v8f wmma_f16(v16h a, v16h b, v8f c) {
  // (neg_a, A, neg_b, B, c_mod, C, reuse_a, reuse_b)
  return __builtin_amdgcn_wmma_f32_16x16x32_f16(false, a, false, b, (short)0, c,
                                                false, false);
}

// ---------------------------------------------------------------- converters
__global__ __launch_bounds__(256) void conv_f32_to_f16(const float* __restrict__ s,
                                                       _Float16* __restrict__ d,
                                                       long long n) {
  long long i = (long long)blockIdx.x * blockDim.x + threadIdx.x;
  if (i < n) d[i] = (_Float16)s[i];
}

// src: (K x N) f32 row-major  ->  dst: (N x K) f16 row-major (i.e. W^T)
__global__ __launch_bounds__(256) void transpose_to_f16(const float* __restrict__ src,
                                                        _Float16* __restrict__ dst,
                                                        int K, int N) {
  long long i = (long long)blockIdx.x * blockDim.x + threadIdx.x;
  long long total = (long long)K * N;
  if (i >= total) return;
  int k = (int)(i / N);
  int n = (int)(i % N);
  dst[(long long)n * K + k] = (_Float16)src[i];
}

// ---------------------------------------------------------------- WMMA GEMM
// C(M x N) = A(M x Kd, f16 row-major) * B(Kd x N)  + bias(N)
// B is supplied transposed: Bt (N x Kd, f16 row-major).
// Wave computes a 32x64 tile; 8 waves/block (4 in M x 2 in N) -> 128x128 tile.
// mode 0: Q  -> outH scattered as (b, h, s, d)  f16
// mode 1: K  -> outF = K_cache (b,s,d) f32, outH = same layout f16
// mode 2: V  -> outF = V_cache (b,s,d) f32, outH = V^T (b, d, s) f16
// mode 3: O  -> outF = final out (M x N) f32
__global__ __launch_bounds__(256) void wmma_gemm_kernel(
    const _Float16* __restrict__ A, const _Float16* __restrict__ Bt,
    const float* __restrict__ bias, float* __restrict__ outF,
    _Float16* __restrict__ outH, int M, int N, int Kd, int srows, int mode) {
  const int wave = threadIdx.x >> 5;
  const int lane = threadIdx.x & 31;
  const int row  = lane & 15;   // A-row / B-col within 16
  const int half = lane >> 4;   // lane half selects K sub-range
  const int wm = wave & 3;      // 4 waves tile M
  const int wn = wave >> 2;     // 2 waves tile N
  const int tileM = blockIdx.y * 128 + wm * 32;
  const int tileN = blockIdx.x * 128 + wn * 64;
  (void)M;

  v8f acc[2][4] = {};
  const _Float16* arow0 = A + (long long)(tileM + row) * Kd;
  const _Float16* arow1 = A + (long long)(tileM + 16 + row) * Kd;

  for (int k0 = 0; k0 < Kd; k0 += 32) {
    // gfx1250 prefetch of next k-tile (global_prefetch_b8)
    __builtin_prefetch(arow0 + k0 + 64, 0, 1);
    __builtin_prefetch(arow1 + k0 + 64, 0, 1);
    // A fragments (16x32): lane half 0 -> K 0-15 ; half 1 -> K 16-31
    v16h af0 = mkv16(*(const v8h*)(arow0 + k0 + half * 16),
                     *(const v8h*)(arow0 + k0 + half * 16 + 8));
    v16h af1 = mkv16(*(const v8h*)(arow1 + k0 + half * 16),
                     *(const v8h*)(arow1 + k0 + half * 16 + 8));
#pragma unroll
    for (int t = 0; t < 4; ++t) {
      // B fragment (32x16): lane col n, contiguous K from Bt row n.
      v16h bf = *(const v16h*)(Bt + (long long)(tileN + t * 16 + row) * Kd +
                               k0 + half * 16);
      acc[0][t] = wmma_f16(af0, bf, acc[0][t]);
      acc[1][t] = wmma_f16(af1, bf, acc[1][t]);
    }
  }

  // Epilogue. C/D layout: VGPR g, lane -> (row = base + g + half*8,
  //                                        col = tileN + t*16 + (lane&15))
#pragma unroll
  for (int rt = 0; rt < 2; ++rt) {
#pragma unroll
    for (int t = 0; t < 4; ++t) {
      const int c = tileN + t * 16 + row;
      const float bv = bias[c];
#pragma unroll
      for (int g = 0; g < 8; ++g) {
        const int r = tileM + rt * 16 + g + half * 8;
        const float v = acc[rt][t][g] + bv;
        if (mode == 3) {
          outF[(long long)r * N + c] = v;
        } else if (mode == 1) {
          outF[(long long)r * N + c] = v;
          outH[(long long)r * N + c] = (_Float16)v;
        } else if (mode == 2) {
          outF[(long long)r * N + c] = v;
          const int bb = r / srows, s = r % srows;
          outH[((long long)bb * N + c) * srows + s] = (_Float16)v;  // V^T
        } else {  // mode 0: Q scatter to (b, h, s, d)
          const int bb = r / srows, s = r % srows;
          const int hh = c >> 7, d = c & 127;
          outH[(((long long)(bb * NHEADS + hh)) * srows + s) * DK + d] =
              (_Float16)v;
        }
      }
    }
  }
}

// ---------------------------------------------------------- flash attention
// Wave owns 16 query rows of one (b, h); loops 32-key blocks over S=2048.
// Qh: (b,h,s,d) f16 ; Kh: (b,s,d) f16 ; Vt: (b,d,s) f16 ; Oh: (b,s,DMODEL) f16
__global__ __launch_bounds__(256) void mqa_attn_kernel(
    const _Float16* __restrict__ Qh, const _Float16* __restrict__ Kh,
    const _Float16* __restrict__ Vt, _Float16* __restrict__ Oh) {
  __shared__ __align__(16) _Float16 plds[8][16 * 32];  // per-wave 16x32 P tile
  const int wave = threadIdx.x >> 5;
  const int lane = threadIdx.x & 31;
  const int row  = lane & 15;
  const int half = lane >> 4;
  const int b = blockIdx.z, h = blockIdx.y;
  const int q0 = (blockIdx.x * 8 + wave) * 16;

  // Preload Q fragments: 16x128 = 4 A-fragments of 16x32.
  const _Float16* qrow =
      Qh + (((long long)(b * NHEADS + h)) * SEQN + q0 + row) * DK;
  v16h qf[4];
#pragma unroll
  for (int kk = 0; kk < 4; ++kk) {
    v8h lo = *(const v8h*)(qrow + kk * 32 + half * 16);
    v8h hi = *(const v8h*)(qrow + kk * 32 + half * 16 + 8);
    qf[kk] = mkv16(lo, hi);
  }

  const _Float16* Kb = Kh + (long long)b * SEQN * DK;
  const _Float16* Vb = Vt + (long long)b * DK * SEQN;
  _Float16* pl = plds[wave];

  float mr[8], lr[8], alph[8];
#pragma unroll
  for (int g = 0; g < 8; ++g) { mr[g] = -1e30f; lr[g] = 0.f; }
  v8f o[8] = {};

  const float scale = 0.08838834764831845f;  // 1/sqrt(128)

  for (int j = 0; j < SEQN; j += 32) {
    // scores for keys [j, j+16) and [j+16, j+32)
    v8f s0 = {}, s1 = {};
#pragma unroll
    for (int kk = 0; kk < 4; ++kk) {
      v16h kf0 = *(const v16h*)(Kb + (long long)(j + row) * DK + kk * 32 + half * 16);
      v16h kf1 = *(const v16h*)(Kb + (long long)(j + 16 + row) * DK + kk * 32 + half * 16);
      s0 = wmma_f16(qf[kk], kf0, s0);
      s1 = wmma_f16(qf[kk], kf1, s1);
    }

    // online softmax: row m = g + half*8 lives in the 16 lanes of this half
#pragma unroll
    for (int g = 0; g < 8; ++g) {
      float a0 = s0[g] * scale, a1 = s1[g] * scale;
      float mx = fmaxf(a0, a1);
      mx = fmaxf(mx, __shfl_xor(mx, 1, 32));
      mx = fmaxf(mx, __shfl_xor(mx, 2, 32));
      mx = fmaxf(mx, __shfl_xor(mx, 4, 32));
      mx = fmaxf(mx, __shfl_xor(mx, 8, 32));
      const float mnew = fmaxf(mr[g], mx);
      const float al = __expf(mr[g] - mnew);
      mr[g] = mnew;
      alph[g] = al;
      const float p0 = __expf(a0 - mnew);
      const float p1 = __expf(a1 - mnew);
      float rs = p0 + p1;
      rs += __shfl_xor(rs, 1, 32);
      rs += __shfl_xor(rs, 2, 32);
      rs += __shfl_xor(rs, 4, 32);
      rs += __shfl_xor(rs, 8, 32);
      lr[g] = lr[g] * al + rs;
      const int rr = g + half * 8;  // query row within tile
      pl[rr * 32 + row] = (_Float16)p0;
      pl[rr * 32 + 16 + row] = (_Float16)p1;
    }

    // rescale output accumulators by alpha (per row g)
#pragma unroll
    for (int dc = 0; dc < 8; ++dc)
#pragma unroll
      for (int g = 0; g < 8; ++g) o[dc][g] *= alph[g];

    // C-layout P -> A-layout P fragment via LDS (wave-private, in-order DS)
    asm volatile("s_wait_dscnt 0" ::: "memory");
    v8h plo = *(const v8h*)(pl + row * 32 + half * 16);
    v8h phi = *(const v8h*)(pl + row * 32 + half * 16 + 8);
    v16h pf = mkv16(plo, phi);

    // O += P (16x32) @ V_block (32x128), using V^T rows for contiguous B frags
#pragma unroll
    for (int dc = 0; dc < 8; ++dc) {
      v16h vf = *(const v16h*)(Vb + (long long)(dc * 16 + row) * SEQN + j + half * 16);
      o[dc] = wmma_f16(pf, vf, o[dc]);
    }
  }

  // normalize and write f16 attention output, layout (b, s, h*128 + d)
#pragma unroll
  for (int g = 0; g < 8; ++g) {
    const float inv = 1.0f / lr[g];
    const int r = q0 + g + half * 8;
    const long long base = ((long long)b * SEQN + r) * DMODEL + h * DK;
#pragma unroll
    for (int dc = 0; dc < 8; ++dc)
      Oh[base + dc * 16 + row] = (_Float16)(o[dc][g] * inv);
  }
}

// ---------------------------------------------------------------- launcher
extern "C" void kernel_launch(void* const* d_in, const int* in_sizes, int n_in,
                              void* d_out, int out_size, void* d_ws,
                              size_t ws_size, hipStream_t stream) {
  (void)in_sizes; (void)n_in; (void)out_size; (void)ws_size;
  const float* x  = (const float*)d_in[0];
  const float* Wq = (const float*)d_in[1];
  const float* bq = (const float*)d_in[2];
  const float* Wk = (const float*)d_in[3];
  const float* bk = (const float*)d_in[4];
  const float* Wv = (const float*)d_in[5];
  const float* bv = (const float*)d_in[6];
  const float* Wo = (const float*)d_in[7];
  const float* bo = (const float*)d_in[8];

  float* out    = (float*)d_out;                                  // (B,S,DMODEL)
  float* kcache = out + (long long)BATCHN * SEQN * DMODEL;        // (B,1,S,DK)
  float* vcache = kcache + (long long)BATCHN * SEQN * DK;         // (B,1,S,DK)

  const long long ROWS = (long long)BATCHN * SEQN;                // 4096
  _Float16* ws = (_Float16*)d_ws;
  _Float16* xh   = ws;                         // 4096*2048
  _Float16* WqT  = xh   + ROWS * DMODEL;       // 2048*2048
  _Float16* WkT  = WqT  + (long long)DMODEL * DMODEL;  // 128*2048
  _Float16* WvT  = WkT  + (long long)DK * DMODEL;
  _Float16* WoT  = WvT  + (long long)DK * DMODEL;      // 2048*2048
  _Float16* Qhs  = WoT  + (long long)DMODEL * DMODEL;  // (B,H,S,DK)
  _Float16* Khs  = Qhs  + ROWS * DMODEL;               // (B,S,DK)
  _Float16* Vts  = Khs  + ROWS * DK;                   // (B,DK,S)
  _Float16* AOs  = Vts  + ROWS * DK;                   // (B,S,DMODEL)

  // 1) precision conversion + weight transposes
  {
    long long n = ROWS * DMODEL;
    conv_f32_to_f16<<<(int)((n + 255) / 256), 256, 0, stream>>>(x, xh, n);
  }
  transpose_to_f16<<<(DMODEL * DMODEL + 255) / 256, 256, 0, stream>>>(Wq, WqT, DMODEL, DMODEL);
  transpose_to_f16<<<(DMODEL * DK + 255) / 256, 256, 0, stream>>>(Wk, WkT, DMODEL, DK);
  transpose_to_f16<<<(DMODEL * DK + 255) / 256, 256, 0, stream>>>(Wv, WvT, DMODEL, DK);
  transpose_to_f16<<<(DMODEL * DMODEL + 255) / 256, 256, 0, stream>>>(Wo, WoT, DMODEL, DMODEL);

  // 2) projections (WMMA GEMMs), block tile 128x128 (8 waves of 32x64)
  dim3 blk(256);
  wmma_gemm_kernel<<<dim3(DMODEL / 128, (int)(ROWS / 128)), blk, 0, stream>>>(
      xh, WqT, bq, nullptr, Qhs, (int)ROWS, DMODEL, DMODEL, SEQN, 0);
  wmma_gemm_kernel<<<dim3(DK / 128, (int)(ROWS / 128)), blk, 0, stream>>>(
      xh, WkT, bk, kcache, Khs, (int)ROWS, DK, DMODEL, SEQN, 1);
  wmma_gemm_kernel<<<dim3(DK / 128, (int)(ROWS / 128)), blk, 0, stream>>>(
      xh, WvT, bv, vcache, Vts, (int)ROWS, DK, DMODEL, SEQN, 2);

  // 3) flash attention (WMMA), 16 queries per wave
  mqa_attn_kernel<<<dim3(SEQN / 128, NHEADS, BATCHN), blk, 0, stream>>>(
      Qhs, Khs, Vts, AOs);

  // 4) output projection
  wmma_gemm_kernel<<<dim3(DMODEL / 128, (int)(ROWS / 128)), blk, 0, stream>>>(
      AOs, WoT, bo, out, nullptr, (int)ROWS, DMODEL, DMODEL, SEQN, 3);
}
